// Net_87497073754244
// MI455X (gfx1250) — matrix operand order repaired and checked by reading, not compiled, not run on hardware
//
#include <hip/hip_runtime.h>
#include <hip/hip_bf16.h>

// ---------------------------------------------------------------------------
// Bipartite GNN (MILP-style) for MI455X / gfx1250.
//
// Strategy:
//  * Per-edge MLPs depend only on x[row]  ->  evaluate per NODE with WMMA
//    (v_wmma_f32_16x16x32_f16), then edges just gather+scale+atomic-scatter.
//  * Weights pre-converted once into exact B-fragment layout (f16) in ws.
//  * Activations kept as f16 shadow (row-major) for direct A-fragment loads.
//  * hj = x[:, :63] implemented by zeroing padded weight row K=63.
//  * All gather/scatter tables fit in 192MB L2 -> edge phase is L2-bound.
// ---------------------------------------------------------------------------

#define N_VAR  40000
#define N_CON  25000
#define N_NODE 65000
#define N_EDGE 400000
#define NPAD   65024   // node rows padded to multiple of 16 (and 64)

typedef __attribute__((ext_vector_type(16))) _Float16 v16h;
typedef __attribute__((ext_vector_type(8)))  _Float16 v8h;
typedef __attribute__((ext_vector_type(8)))  float    v8f;

// ---- WMMA fragment helpers -------------------------------------------------
// A (16x32 f16): lane L -> row = L&15, hi = L>>4.
//   elems 0..7  : K = kb + 8*hi + e
//   elems 8..15 : K = kb + 16 + 8*hi + (e-8)
__device__ __forceinline__ v16h wmma_load_a(const _Float16* rowPtr, int kb, int hi) {
  v8h lo = *(const v8h*)(rowPtr + kb + 8 * hi);
  v8h hi8 = *(const v8h*)(rowPtr + kb + 16 + 8 * hi);
  v16h r;
#pragma unroll
  for (int i = 0; i < 8; ++i) { r[i] = lo[i]; r[i + 8] = hi8[i]; }
  return r;
}

// B (32x16 f16) pre-packed: frag[((kt*Nt+nt)*32 + lane)*16 + e]
__device__ __forceinline__ v16h wmma_load_b(const _Float16* frag, int kt, int nt,
                                            int Nt, int lane) {
  return *(const v16h*)(frag + (size_t)(((kt * Nt + nt) * 32) + lane) * 16);
}

__device__ __forceinline__ v8f wmma_mac(v16h a, v16h b, v8f c) {
  return __builtin_amdgcn_wmma_f32_16x16x32_f16(false, a, false, b, (short)0, c,
                                                false, false);
}

__device__ __forceinline__ float frelu(float v) { return v > 0.f ? v : 0.f; }

// ---- weight -> B-fragment conversion (run once) ----------------------------
__global__ void k_wfrag(const float* __restrict__ src, int fanin, int fanout,
                        int Kt, int Nt, _Float16* __restrict__ dst) {
  int t = blockIdx.x * blockDim.x + threadIdx.x;
  int total = Kt * Nt * 512;
  if (t >= total) return;
  int e = t & 15;
  int lane = (t >> 4) & 31;
  int f = t >> 9;
  int nt = f % Nt;
  int kt = f / Nt;
  int K = kt * 32 + (e < 8 ? (lane >> 4) * 8 + e : 16 + (lane >> 4) * 8 + (e - 8));
  int N = nt * 16 + (lane & 15);
  float val = (K < fanin && N < fanout) ? src[(size_t)K * fanout + N] : 0.f;
  dst[t] = (_Float16)val;
}

// ---- zero helpers ----------------------------------------------------------
__global__ void k_zero(float* __restrict__ p, long n) {
  long t = (long)blockIdx.x * blockDim.x + threadIdx.x;
  if (t < n) p[t] = 0.f;
}
__global__ void k_zero_h(_Float16* __restrict__ p, long n) {
  long t = (long)blockIdx.x * blockDim.x + threadIdx.x;
  if (t < n) p[t] = (_Float16)0.f;
}

// ---- input embedding: 2 -> 61 -> 61, concat raw(2) + zero(1) ---------------
__global__ void k_embed(const float* __restrict__ feat, int n, int nodeOff,
                        const float* __restrict__ W1, const float* __restrict__ b1,
                        const float* __restrict__ W2, const float* __restrict__ b2,
                        float* __restrict__ x, _Float16* __restrict__ xh,
                        _Float16* __restrict__ xsvar) {
  __shared__ float sW1[2 * 61];
  __shared__ float sb1[61];
  __shared__ float sW2[61 * 61];
  __shared__ float sb2[61];
  for (int i = threadIdx.x; i < 122; i += blockDim.x) sW1[i] = W1[i];
  for (int i = threadIdx.x; i < 61; i += blockDim.x) { sb1[i] = b1[i]; sb2[i] = b2[i]; }
  for (int i = threadIdx.x; i < 3721; i += blockDim.x) sW2[i] = W2[i];
  __syncthreads();
  int i = blockIdx.x * blockDim.x + threadIdx.x;
  if (i >= n) return;
  float f0 = feat[2 * i], f1 = feat[2 * i + 1];
  float hid[61];
  for (int h = 0; h < 61; ++h)
    hid[h] = frelu(f0 * sW1[h] + f1 * sW1[61 + h] + sb1[h]);
  int node = nodeOff + i;
  float* xr = x + (size_t)node * 64;
  _Float16* xhr = xh + (size_t)node * 64;
  _Float16* zr = (nodeOff == 0) ? (xsvar + (size_t)node * 448) : (_Float16*)0;
  for (int o = 0; o < 61; ++o) {
    float s = sb2[o];
    for (int h = 0; h < 61; ++h) s += hid[h] * sW2[h * 61 + o];
    xr[o] = s; xhr[o] = (_Float16)s;
    if (zr) zr[o] = (_Float16)s;
  }
  xr[61] = f0; xr[62] = f1; xr[63] = 0.f;
  xhr[61] = (_Float16)f0; xhr[62] = (_Float16)f1; xhr[63] = (_Float16)0.f;
  if (zr) { zr[61] = (_Float16)f0; zr[62] = (_Float16)f1; zr[63] = (_Float16)0.f; }
}

// ---- degree + per-edge precompute ------------------------------------------
__global__ void k_deg(const int* __restrict__ erow, float* __restrict__ deg) {
  int e = blockIdx.x * blockDim.x + threadIdx.x;
  if (e < N_EDGE) atomicAdd(&deg[erow[e]], 1.0f);
}
__global__ void k_edgepre(const int* __restrict__ erow, const int* __restrict__ etyp,
                          const float* __restrict__ deg, const float* __restrict__ ef,
                          float* __restrict__ enorm, float* __restrict__ em63) {
  int e = blockIdx.x * blockDim.x + threadIdx.x;
  if (e >= N_EDGE) return;
  int r = erow[e];
  float nm = 1.0f / deg[r];
  float ce = ef[r];
  enorm[e] = nm;
  em63[e] = (etyp[e] == 0) ? ce : nm * ce;  // type0: htv*ce ; type1: norm*ce*xlast
}

// ---- per-node MLPs (cons 63->63->63, vars 63->63->63, htv 64->63->1) -------
// msgA[node][0..62]=cons_out, msgA[node][63]=htv_out
// msgB[node][0..62]=vars_out, msgB[node][63]=x[node][63]
__global__ void __launch_bounds__(128)
k_node_mlp(const _Float16* __restrict__ xh, const float* __restrict__ x,
           const _Float16* __restrict__ fConsW1, const _Float16* __restrict__ fVarsW1,
           const _Float16* __restrict__ fHtvW1,  const _Float16* __restrict__ fConsW2,
           const _Float16* __restrict__ fVarsW2, const _Float16* __restrict__ fHtvW2,
           const float* __restrict__ consb1, const float* __restrict__ consb2,
           const float* __restrict__ varsb1, const float* __restrict__ varsb2,
           const float* __restrict__ htvb1,  const float* __restrict__ htvb2,
           float* __restrict__ msgA, float* __restrict__ msgB) {
  __shared__ __align__(16) _Float16 hidLds[4][16 * 192];
  int wave = threadIdx.x >> 5;
  int lane = threadIdx.x & 31;
  int tile = blockIdx.x * 4 + wave;
  int nodeBase = tile * 16;
  if (nodeBase >= N_NODE) return;
  int hi = lane >> 4, ln = lane & 15;

  const _Float16* arow = xh + (size_t)(nodeBase + ln) * 64;
  v16h a0 = wmma_load_a(arow, 0, hi);
  v16h a1 = wmma_load_a(arow, 32, hi);
  _Float16* hl = hidLds[wave];

  // GEMM1: hidden = relu(x @ W1pad + b1pad) for the 3 MLPs (192 channels)
  const _Float16* fw1[3] = {fConsW1, fVarsW1, fHtvW1};
  const float* fb1[3] = {consb1, varsb1, htvb1};
#pragma unroll
  for (int m = 0; m < 3; ++m) {
#pragma unroll
    for (int nt = 0; nt < 4; ++nt) {
      v8f acc = {};
      acc = wmma_mac(a0, wmma_load_b(fw1[m], 0, nt, 4, lane), acc);
      acc = wmma_mac(a1, wmma_load_b(fw1[m], 1, nt, 4, lane), acc);
      int ch = nt * 16 + ln;
      float bias = (ch < 63) ? fb1[m][ch] : 0.f;
#pragma unroll
      for (int v = 0; v < 8; ++v) {
        int node = v + 8 * hi;
        hl[node * 192 + m * 64 + ch] = (_Float16)frelu(acc[v] + bias);
      }
    }
  }
  asm volatile("s_wait_dscnt 0" ::: "memory");

  // GEMM2: transpose through LDS, project to outputs
  const _Float16* a2row = hl + ln * 192;
  v16h ac0 = wmma_load_a(a2row, 0, hi);
  v16h ac1 = wmma_load_a(a2row, 32, hi);
  v16h av0 = wmma_load_a(a2row + 64, 0, hi);
  v16h av1 = wmma_load_a(a2row + 64, 32, hi);
  v16h ah0 = wmma_load_a(a2row + 128, 0, hi);
  v16h ah1 = wmma_load_a(a2row + 128, 32, hi);

#pragma unroll
  for (int nt = 0; nt < 4; ++nt) {  // cons -> msgA[:, 0..62]
    v8f acc = {};
    acc = wmma_mac(ac0, wmma_load_b(fConsW2, 0, nt, 4, lane), acc);
    acc = wmma_mac(ac1, wmma_load_b(fConsW2, 1, nt, 4, lane), acc);
    int ch = nt * 16 + ln;
    if (ch < 63) {
      float bias = consb2[ch];
#pragma unroll
      for (int v = 0; v < 8; ++v)
        msgA[(size_t)(nodeBase + v + 8 * hi) * 64 + ch] = acc[v] + bias;
    }
  }
#pragma unroll
  for (int nt = 0; nt < 4; ++nt) {  // vars -> msgB[:, 0..62]
    v8f acc = {};
    acc = wmma_mac(av0, wmma_load_b(fVarsW2, 0, nt, 4, lane), acc);
    acc = wmma_mac(av1, wmma_load_b(fVarsW2, 1, nt, 4, lane), acc);
    int ch = nt * 16 + ln;
    if (ch < 63) {
      float bias = varsb2[ch];
#pragma unroll
      for (int v = 0; v < 8; ++v)
        msgB[(size_t)(nodeBase + v + 8 * hi) * 64 + ch] = acc[v] + bias;
    }
  }
  {  // htv -> msgA[:, 63]
    v8f acc = {};
    acc = wmma_mac(ah0, wmma_load_b(fHtvW2, 0, 0, 1, lane), acc);
    acc = wmma_mac(ah1, wmma_load_b(fHtvW2, 1, 0, 1, lane), acc);
    if (ln == 0) {
      float bias = htvb2[0];
#pragma unroll
      for (int v = 0; v < 8; ++v)
        msgA[(size_t)(nodeBase + v + 8 * hi) * 64 + 63] = acc[v] + bias;
    }
  }
  if (lane < 16)  // xlast copy -> msgB[:, 63]
    msgB[(size_t)(nodeBase + lane) * 64 + 63] = x[(size_t)(nodeBase + lane) * 64 + 63];
}

// ---- edge scatter: one wave per edge, 2 channels per lane ------------------
__global__ void k_edge(const int* __restrict__ erow, const int* __restrict__ ecol,
                       const int* __restrict__ etyp, const float* __restrict__ enorm,
                       const float* __restrict__ em63, const float* __restrict__ msgA,
                       const float* __restrict__ msgB, float* __restrict__ aggr) {
  int gt = blockIdx.x * blockDim.x + threadIdx.x;
  int e = gt >> 5;
  if (e >= N_EDGE) return;
  int lane = gt & 31;
  int r = erow[e], c = ecol[e], t = etyp[e];
  float nm = enorm[e], m63 = em63[e];
  const float* sr = ((t == 0) ? msgA : msgB) + (size_t)r * 64;
  float* dr = aggr + (size_t)c * 64;
  atomicAdd(dr + lane, nm * sr[lane]);
  float s1 = sr[lane + 32];
  atomicAdd(dr + lane + 32, (lane == 31) ? (m63 * s1) : (nm * s1));
}

// ---- bias + rhs correction + relu + pack f16 shadows -----------------------
__global__ void k_finish(const float* __restrict__ aggr, const float* __restrict__ bias,
                         const float* __restrict__ rhs, float* __restrict__ x,
                         _Float16* __restrict__ xh, _Float16* __restrict__ xsvar,
                         int layer) {
  long t = (long)blockIdx.x * blockDim.x + threadIdx.x;
  if (t >= (long)N_NODE * 64) return;
  int node = (int)(t >> 6);
  int c = (int)(t & 63);
  float a = aggr[t] + bias[c];
  if (c == 63 && node >= N_VAR) a -= rhs[node - N_VAR];
  float v = frelu(a);
  x[t] = v;
  xh[t] = (_Float16)v;
  if (node < N_VAR) xsvar[(size_t)node * 448 + (size_t)(layer + 1) * 64 + c] = (_Float16)v;
}

// ---- output head: 448 -> 64 -> 64 -> 64 -> 1 over var rows -----------------
__device__ __forceinline__ void head_gemm64(const _Float16* src, const _Float16* frag,
                                            const float* bias, _Float16* dst, int lane) {
  int hi = lane >> 4, ln = lane & 15;
  const _Float16* arow = src + ln * 64;
  v16h a0 = wmma_load_a(arow, 0, hi);
  v16h a1 = wmma_load_a(arow, 32, hi);
#pragma unroll
  for (int nt = 0; nt < 4; ++nt) {
    v8f acc = {};
    acc = wmma_mac(a0, wmma_load_b(frag, 0, nt, 4, lane), acc);
    acc = wmma_mac(a1, wmma_load_b(frag, 1, nt, 4, lane), acc);
    int ch = nt * 16 + ln;
    float b = bias[ch];
#pragma unroll
    for (int v = 0; v < 8; ++v)
      dst[(v + 8 * hi) * 64 + ch] = (_Float16)frelu(acc[v] + b);
  }
  asm volatile("s_wait_dscnt 0" ::: "memory");
}

__global__ void __launch_bounds__(128)
k_head(const _Float16* __restrict__ xsvar,
       const _Float16* __restrict__ f1, const float* __restrict__ b1,
       const _Float16* __restrict__ f2, const float* __restrict__ b2,
       const _Float16* __restrict__ f3, const float* __restrict__ b3,
       const _Float16* __restrict__ f4, const float* __restrict__ b4,
       float* __restrict__ out) {
  __shared__ __align__(16) _Float16 hA[4][16 * 64];
  __shared__ __align__(16) _Float16 hB[4][16 * 64];
  int wave = threadIdx.x >> 5;
  int lane = threadIdx.x & 31;
  int tile = blockIdx.x * 4 + wave;
  int rowBase = tile * 16;
  if (rowBase >= N_VAR) return;
  int hi = lane >> 4, ln = lane & 15;
  _Float16* HA = hA[wave];
  _Float16* HB = hB[wave];

  // fc1: K=448 (14 k-steps), N=64, relu -> HA
  const _Float16* arow = xsvar + (size_t)(rowBase + ln) * 448;
  v8f acc[4] = {};
#pragma unroll
  for (int kt = 0; kt < 14; ++kt) {
    v16h a = wmma_load_a(arow, 32 * kt, hi);
#pragma unroll
    for (int nt = 0; nt < 4; ++nt)
      acc[nt] = wmma_mac(a, wmma_load_b(f1, kt, nt, 4, lane), acc[nt]);
  }
#pragma unroll
  for (int nt = 0; nt < 4; ++nt) {
    int ch = nt * 16 + ln;
    float b = b1[ch];
#pragma unroll
    for (int v = 0; v < 8; ++v)
      HA[(v + 8 * hi) * 64 + ch] = (_Float16)frelu(acc[nt][v] + b);
  }
  asm volatile("s_wait_dscnt 0" ::: "memory");

  head_gemm64(HA, f2, b2, HB, lane);  // fc2 + relu
  head_gemm64(HB, f3, b3, HA, lane);  // fc3 + relu

  // fc4: 64 -> 1
  {
    const _Float16* a4row = HA + ln * 64;
    v16h a0 = wmma_load_a(a4row, 0, hi);
    v16h a1 = wmma_load_a(a4row, 32, hi);
    v8f acc4 = {};
    acc4 = wmma_mac(a0, wmma_load_b(f4, 0, 0, 1, lane), acc4);
    acc4 = wmma_mac(a1, wmma_load_b(f4, 1, 0, 1, lane), acc4);
    if (ln == 0) {
      float bb = b4[0];
#pragma unroll
      for (int v = 0; v < 8; ++v) out[rowBase + v + 8 * hi] = acc4[v] + bb;
    }
  }
}

// ---------------------------------------------------------------------------
extern "C" void kernel_launch(void* const* d_in, const int* in_sizes, int n_in,
                              void* d_out, int out_size, void* d_ws, size_t ws_size,
                              hipStream_t stream) {
  (void)in_sizes; (void)n_in; (void)out_size; (void)ws_size;
  const float* vf = (const float*)d_in[0];
  const float* cf = (const float*)d_in[1];
  const int* erow = (const int*)d_in[2];
  const int* ecol = erow + N_EDGE;
  const int* etyp = (const int*)d_in[3];
  const float* ef = (const float*)d_in[4];
  const float* rhs = (const float*)d_in[5];
  // params flattened in JAX pytree (alphabetical dict-key) order:
  // 8..11 con_mlp(W1,b1,W2,b2); 12+13l convs[l]{bias,consW1,consb1,consW2,consb2,
  // htvW1,htvb1,htvW2,htvb2,varsW1,varsb1,varsW2,varsb2}; 90..97 fc1..fc4(W,b);
  // 98..101 var_mlp(W1,b1,W2,b2)
  auto F = [&](int i) { return (const float*)d_in[i]; };

  char* ws = (char*)d_ws;
  size_t off = 0;
  auto alloc = [&](size_t bytes) -> char* {
    off = (off + 255) & ~(size_t)255;
    char* p = ws + off;
    off += bytes;
    return p;
  };
  float* x = (float*)alloc((size_t)NPAD * 64 * 4);
  _Float16* xh = (_Float16*)alloc((size_t)NPAD * 64 * 2);
  float* aggr = (float*)alloc((size_t)NPAD * 64 * 4);
  float* msgA = (float*)alloc((size_t)NPAD * 64 * 4);
  float* msgB = (float*)alloc((size_t)NPAD * 64 * 4);
  float* deg = (float*)alloc((size_t)N_NODE * 4);
  float* enorm = (float*)alloc((size_t)N_EDGE * 4);
  float* em63 = (float*)alloc((size_t)N_EDGE * 4);
  _Float16* xsvar = (_Float16*)alloc((size_t)N_VAR * 448 * 2);

  auto fragAlloc = [&](int Kt, int Nt) -> _Float16* {
    return (_Float16*)alloc((size_t)Kt * Nt * 512 * 2);
  };
  _Float16 *fConsW1[6], *fVarsW1[6], *fHtvW1[6], *fConsW2[6], *fVarsW2[6], *fHtvW2[6];
  for (int l = 0; l < 6; ++l) {
    fConsW1[l] = fragAlloc(2, 4);
    fVarsW1[l] = fragAlloc(2, 4);
    fHtvW1[l] = fragAlloc(2, 4);
    fConsW2[l] = fragAlloc(2, 4);
    fVarsW2[l] = fragAlloc(2, 4);
    fHtvW2[l] = fragAlloc(2, 1);
  }
  _Float16* fFC1 = fragAlloc(14, 4);
  _Float16* fFC2 = fragAlloc(2, 4);
  _Float16* fFC3 = fragAlloc(2, 4);
  _Float16* fFC4 = fragAlloc(2, 1);

  auto wconv = [&](const float* src, int fanin, int fanout, int Kt, int Nt,
                   _Float16* dst) {
    int n = Kt * Nt * 512;
    k_wfrag<<<(n + 255) / 256, 256, 0, stream>>>(src, fanin, fanout, Kt, Nt, dst);
  };
  for (int l = 0; l < 6; ++l) {
    int b = 12 + 13 * l;
    wconv(F(b + 1), 63, 63, 2, 4, fConsW1[l]);   // consW1 (row63 zero => hj)
    wconv(F(b + 9), 63, 63, 2, 4, fVarsW1[l]);   // varsW1
    wconv(F(b + 5), 64, 63, 2, 4, fHtvW1[l]);    // htvW1 (full 64-ch input)
    wconv(F(b + 3), 63, 63, 2, 4, fConsW2[l]);   // consW2
    wconv(F(b + 11), 63, 63, 2, 4, fVarsW2[l]);  // varsW2
    wconv(F(b + 7), 63, 1, 2, 1, fHtvW2[l]);     // htvW2 (63 -> 1)
  }
  wconv(F(90), 448, 64, 14, 4, fFC1);
  wconv(F(92), 64, 64, 2, 4, fFC2);
  wconv(F(94), 64, 64, 2, 4, fFC3);
  wconv(F(96), 64, 1, 2, 1, fFC4);

  // degrees and per-edge precompute
  k_zero<<<(N_NODE + 255) / 256, 256, 0, stream>>>(deg, N_NODE);
  k_deg<<<(N_EDGE + 255) / 256, 256, 0, stream>>>(erow, deg);
  k_edgepre<<<(N_EDGE + 255) / 256, 256, 0, stream>>>(erow, etyp, deg, ef, enorm, em63);

  // zero padded node rows (read by tail WMMA tile)
  long padN = (long)(NPAD - N_NODE) * 64;
  k_zero<<<(int)((padN + 255) / 256), 256, 0, stream>>>(x + (size_t)N_NODE * 64, padN);
  k_zero_h<<<(int)((padN + 255) / 256), 256, 0, stream>>>(xh + (size_t)N_NODE * 64, padN);

  // input embeddings (x0 + f16 shadows + xs slice 0)
  k_embed<<<(N_VAR + 255) / 256, 256, 0, stream>>>(vf, N_VAR, 0, F(98), F(99), F(100),
                                                   F(101), x, xh, xsvar);
  k_embed<<<(N_CON + 255) / 256, 256, 0, stream>>>(cf, N_CON, N_VAR, F(8), F(9), F(10),
                                                   F(11), x, xh, xsvar);

  int nTiles = (N_NODE + 15) / 16;       // 4063
  int nodeBlocks = (nTiles + 3) / 4;     // 1016
  long aggrN = (long)NPAD * 64;
  for (int l = 0; l < 6; ++l) {
    int b = 12 + 13 * l;
    k_zero<<<(int)((aggrN + 255) / 256), 256, 0, stream>>>(aggr, aggrN);
    k_node_mlp<<<nodeBlocks, 128, 0, stream>>>(
        xh, x, fConsW1[l], fVarsW1[l], fHtvW1[l], fConsW2[l], fVarsW2[l], fHtvW2[l],
        F(b + 2), F(b + 4), F(b + 10), F(b + 12), F(b + 6), F(b + 8), msgA, msgB);
    k_edge<<<(N_EDGE * 32) / 256, 256, 0, stream>>>(erow, ecol, etyp, enorm, em63,
                                                    msgA, msgB, aggr);
    long finN = (long)N_NODE * 64;
    k_finish<<<(int)((finN + 255) / 256), 256, 0, stream>>>(aggr, F(b + 0), rhs, x, xh,
                                                            xsvar, l);
  }

  // readout head over 2500 tiles of 16 var rows
  k_head<<<(N_VAR / 16 + 3) / 4, 128, 0, stream>>>(xsvar, fFC1, F(91), fFC2, F(93),
                                                   fFC3, F(95), fFC4, F(97),
                                                   (float*)d_out);
}
